// SLUSeqTagger_53085795778919
// MI455X (gfx1250) — compile-verified, matrix operand order checked by reading, not run on hardware
//
#include <hip/hip_runtime.h>
#include <hip/hip_bf16.h>
#include <math.h>

// Problem constants (from reference)
#define BB 64
#define TT 512
#define HH 256
#define DD 512          // 2*H
#define MS 64           // MAX_SLOTS
#define ML 8            // MAX_LEN
#define NS 64           // N_SLOT_TYPES
#define KK 768          // D + H  (fused [x|h] contraction dim)
#define GATES 1024      // 4*H

// dynamic LDS layout (bytes)
#define AX_BYTES   (2 * 16 * KK * 2)     // 49152: A tile [dir][16][768] f16
#define HB_BYTES   (2 * 16 * HH * 2)     // 16384: h double buffer [dir][16][256] f16
#define LEN_OFF    (AX_BYTES + HB_BYTES)
#define TOK_OFF    (LEN_OFF + 16 * 4)
#define SMEM_BYTES (TOK_OFF + 16 * ML * 4)

typedef _Float16 v4h  __attribute__((ext_vector_type(4)));
typedef _Float16 v8h  __attribute__((ext_vector_type(8)));
typedef _Float16 v16h __attribute__((ext_vector_type(16)));
typedef float    v4f  __attribute__((ext_vector_type(4)));
typedef float    v8f  __attribute__((ext_vector_type(8)));

__device__ __forceinline__ v16h mkfrag(const _Float16* p) {
    // two 16B chunks: k = [0..7] and [16..23] relative to p (per ISA 16-bit A/B layout)
    v8h lo = *(const v8h*)(p);
    v8h hi = *(const v8h*)(p + 16);
    return __builtin_shufflevector(lo, hi, 0,1,2,3,4,5,6,7,8,9,10,11,12,13,14,15);
}

__device__ __forceinline__ float sigm(float x) { return 1.0f / (1.0f + __expf(-x)); }

// ---------------------------------------------------------------------------
// Kernel 0: BIO -> span token tables.  tok[b*MS+s][ML], lens[b*MS+s]
// ---------------------------------------------------------------------------
__global__ void pack_spans_kernel(const int* __restrict__ bio,
                                  int* __restrict__ tok, int* __restrict__ lens) {
    int b = blockIdx.x;
    if (threadIdx.x != 0) return;
    int* lrow = lens + b * MS;
    int* trow = tok + b * MS * ML;
    for (int s = 0; s < MS; ++s) lrow[s] = 0;
    const int* brow = bio + b * TT;
    int nB = 0, cur = -1, cnt = 0;
    for (int t = 0; t < TT; ++t) {
        int v = brow[t];
        if (v == 1) { cur = nB++; cnt = 0; }
        if (v == 1 || (v == 2 && cur >= 0)) {
            if (cur >= 0 && cur < MS && cnt < ML) {
                trow[cur * ML + cnt] = t;
                lrow[cur] = cnt + 1;
            }
            cnt++;
        }
    }
}

// ---------------------------------------------------------------------------
// Kernel 0b: convert weights to f16.
// Wf16 layout: [dir][gate_row n=0..1023][k=0..767], k<512 -> W_ih, else W_hh
// ---------------------------------------------------------------------------
__global__ void cvt_weights_kernel(const float* __restrict__ Wih_f, const float* __restrict__ Whh_f,
                                   const float* __restrict__ Wih_b, const float* __restrict__ Whh_b,
                                   const float* __restrict__ slot,
                                   _Float16* __restrict__ Wf16, _Float16* __restrict__ slotF16) {
    int i = blockIdx.x * blockDim.x + threadIdx.x;
    const int WE = 2 * GATES * KK;
    if (i < WE) {
        int dir = i / (GATES * KK);
        int r = (i / KK) % GATES;
        int k = i % KK;
        const float* ih = dir ? Wih_b : Wih_f;
        const float* hh = dir ? Whh_b : Whh_f;
        float v = (k < DD) ? ih[r * DD + k] : hh[r * HH + (k - DD)];
        Wf16[i] = (_Float16)v;
    } else if (i < WE + NS * DD) {
        int j = i - WE;
        slotF16[j] = (_Float16)slot[j];
    }
}

// ---------------------------------------------------------------------------
// Kernel 1: fused gather + bidirectional slot-LSTM + sum-pool + scores.
// One workgroup (8 waves, 256 thr) owns 16 samples for the whole recurrence.
// ---------------------------------------------------------------------------
__global__ __launch_bounds__(256) void slot_lstm_fused_kernel(
    const float* __restrict__ repr,      // [B][T][D]
    const float* __restrict__ bias_f,    // [4H]
    const float* __restrict__ bias_b,    // [4H]
    const int*   __restrict__ tok,       // [B*MS][ML]
    const int*   __restrict__ lens,      // [B*MS]
    const _Float16* __restrict__ Wf16,   // [2][1024][768]
    const _Float16* __restrict__ slotF16,// [64][512]
    float* __restrict__ out)             // [B*MS][NS]
{
    extern __shared__ char smem[];
    _Float16 (*Ax)[16][KK] = (_Float16(*)[16][KK])(smem);             // [2][16][768]
    _Float16 (*Hb)[16][HH] = (_Float16(*)[16][HH])(smem + AX_BYTES);  // [2][16][256]
    int* sLen = (int*)(smem + LEN_OFF);
    int (*sTok)[ML] = (int(*)[ML])(smem + TOK_OFF);

    const int tid  = threadIdx.x;
    const int wave = tid >> 5;
    const int lane = tid & 31;
    const int nloc = lane & 15;
    const int halfsel = lane >> 4;
    const int base = blockIdx.x * 16;    // first sample (b*MS+slot flat index)

    if (tid < 16) sLen[tid] = lens[base + tid];
    if (tid < 16 * ML) { int s = tid >> 3, r = tid & 7; sTok[s][r] = tok[(base + s) * ML + r]; }
    // zero initial h double-buffer (vectorized: 16B stores)
    for (int e = tid; e < 2 * 16 * (HH / 8); e += 256) {
        int dir = e >> 9, s = (e >> 5) & 15, k8 = (e & 31) << 3;
        v8h z = (v8h)(_Float16)0.0f;
        *(v8h*)&Hb[dir][s][k8] = z;
    }
    __syncthreads();

    // loop-invariant per-lane values
    int lenr[8];
    #pragma unroll
    for (int j = 0; j < 8; ++j) lenr[j] = sLen[j + halfsel * 8];

    float biasv[4][4];
    #pragma unroll
    for (int q = 0; q < 4; ++q) {
        const int task = wave + q * 8;
        const int dir = task >> 4;
        const int ht = task & 15;
        const float* bias = dir ? bias_b : bias_f;
        #pragma unroll
        for (int g = 0; g < 4; ++g) biasv[q][g] = bias[g * HH + ht * 16 + nloc];
    }

    v8f cst[4], pool[4];
    #pragma unroll
    for (int q = 0; q < 4; ++q)
        #pragma unroll
        for (int j = 0; j < 8; ++j) { cst[q][j] = 0.0f; pool[q][j] = 0.0f; }

    for (int t = 0; t < ML; ++t) {
        // ---- phase A (vectorized): gather x_t (fwd) / x_{len-1-t} (bwd), f32x4 -> f16x4
        for (int e = tid; e < 2 * 16 * (DD / 4); e += 256) {   // 4096 float4 chunks
            int dir = e >> 11;            // 16*128 chunks per dir
            int s = (e >> 7) & 15;
            int k4 = (e & 127) << 2;      // 0..508 step 4
            int len = sLen[s];
            v4h h4;
            if (t < len) {
                int ti = (dir == 0) ? sTok[s][t] : sTok[s][len - 1 - t];
                int b = (base + s) >> 6;
                v4f v = *(const v4f*)(repr + ((size_t)(b * TT + ti)) * DD + k4);
                h4[0] = (_Float16)v[0]; h4[1] = (_Float16)v[1];
                h4[2] = (_Float16)v[2]; h4[3] = (_Float16)v[3];
            } else {
                h4 = (v4h)(_Float16)0.0f;
            }
            *(v4h*)&Ax[dir][s][k4] = h4;
        }
        // copy Hb -> A h-columns (16B LDS-to-LDS moves)
        for (int e = tid; e < 2 * 16 * (HH / 8); e += 256) {   // 1024 chunks
            int dir = e >> 9, s = (e >> 5) & 15, k8 = (e & 31) << 3;
            *(v8h*)&Ax[dir][s][DD + k8] = *(const v8h*)&Hb[dir][s][k8];
        }
        __syncthreads();

        // ---- per task: gate GEMM then immediate cell update (only 4 accs live)
        #pragma unroll
        for (int q = 0; q < 4; ++q) {
            const int task = wave + q * 8;
            const int dir = task >> 4;
            const int ht = task & 15;
            const _Float16* Arow = &Ax[dir][nloc][0];
            const _Float16* Wbase = Wf16 + (size_t)dir * GATES * KK + (size_t)(ht * 16 + nloc) * KK;

            v8f acc[4];
            #pragma unroll
            for (int g = 0; g < 4; ++g)
                #pragma unroll
                for (int j = 0; j < 8; ++j) acc[g][j] = 0.0f;

            for (int kb = 0; kb < KK / 32; ++kb) {
                const int k0 = kb * 32 + halfsel * 8;
                v16h af = mkfrag(Arow + k0);
                #pragma unroll
                for (int g = 0; g < 4; ++g) {
                    v16h bf = mkfrag(Wbase + (size_t)g * HH * KK + k0);
                    acc[g] = __builtin_amdgcn_wmma_f32_16x16x32_f16(
                        false, af, false, bf, (short)0, acc[g], false, false);
                }
            }

            // epilogue: LSTM cell update, masked pooling, h_t -> Hb (not Ax!)
            #pragma unroll
            for (int j = 0; j < 8; ++j) {
                const int m = j + halfsel * 8;
                float iv = sigm(acc[0][j] + biasv[q][0]);
                float fv = sigm(acc[1][j] + biasv[q][1]);
                float gv = tanhf(acc[2][j] + biasv[q][2]);
                float ov = sigm(acc[3][j] + biasv[q][3]);
                float c = fv * cst[q][j] + iv * gv;
                cst[q][j] = c;
                float h = ov * tanhf(c);
                if (t < lenr[j]) pool[q][j] += h;
                Hb[dir][m][ht * 16 + nloc] = (_Float16)h;
            }
        }
        __syncthreads();   // Ax reads done; Hb writes visible for next phase A
    }

    // ---- pooled tile -> LDS (reuse Ax storage as P[16][512] f16)
    _Float16* P = &Ax[0][0][0];
    #pragma unroll
    for (int q = 0; q < 4; ++q) {
        const int task = wave + q * 8;
        const int dir = task >> 4;
        const int ht = task & 15;
        #pragma unroll
        for (int j = 0; j < 8; ++j) {
            const int m = j + halfsel * 8;
            P[m * DD + dir * HH + ht * 16 + nloc] = (_Float16)pool[q][j];
        }
    }
    __syncthreads();

    // ---- scores: [16 x 512] @ slot_emb^T [512 x 64], waves 0..3 take one N-tile
    if (wave < 4) {
        v8f sacc;
        #pragma unroll
        for (int j = 0; j < 8; ++j) sacc[j] = 0.0f;
        const _Float16* Prow = P + nloc * DD;
        const _Float16* Srow = slotF16 + (size_t)(wave * 16 + nloc) * DD;
        for (int kb = 0; kb < DD / 32; ++kb) {
            const int k0 = kb * 32 + halfsel * 8;
            v16h af = mkfrag(Prow + k0);
            v16h bf = mkfrag(Srow + k0);
            sacc = __builtin_amdgcn_wmma_f32_16x16x32_f16(
                false, af, false, bf, (short)0, sacc, false, false);
        }
        #pragma unroll
        for (int j = 0; j < 8; ++j) {
            float v = (lenr[j] > 0) ? sacc[j] : 0.0f;
            const int m = j + halfsel * 8;
            out[(size_t)(base + m) * NS + wave * 16 + nloc] = v;
        }
    }
}

// ---------------------------------------------------------------------------
extern "C" void kernel_launch(void* const* d_in, const int* in_sizes, int n_in,
                              void* d_out, int out_size, void* d_ws, size_t ws_size,
                              hipStream_t stream) {
    const float* repr  = (const float*)d_in[0];
    const float* Wih_f = (const float*)d_in[1];
    const float* Whh_f = (const float*)d_in[2];
    const float* b_f   = (const float*)d_in[3];
    const float* Wih_b = (const float*)d_in[4];
    const float* Whh_b = (const float*)d_in[5];
    const float* b_b   = (const float*)d_in[6];
    const float* slot  = (const float*)d_in[7];
    const int*   bio   = (const int*)d_in[8];
    float* out = (float*)d_out;

    char* ws = (char*)d_ws;
    int* tok  = (int*)ws;                                   // 64*64*8*4   = 128KB
    int* lens = (int*)(ws + 131072);                        // 64*64*4     = 16KB
    _Float16* Wf16    = (_Float16*)(ws + 131072 + 16384);   // 2*1024*768*2 = 3MB
    _Float16* slotF16 = (_Float16*)((char*)Wf16 + (size_t)2 * GATES * KK * 2);

    pack_spans_kernel<<<BB, 32, 0, stream>>>(bio, tok, lens);

    const int tot = 2 * GATES * KK + NS * DD;
    cvt_weights_kernel<<<(tot + 255) / 256, 256, 0, stream>>>(
        Wih_f, Whh_f, Wih_b, Whh_b, slot, Wf16, slotF16);

    slot_lstm_fused_kernel<<<(BB * MS) / 16, 256, SMEM_BYTES, stream>>>(
        repr, b_f, b_b, tok, lens, Wf16, slotF16, out);
}
